// Attention_12799002542545
// MI455X (gfx1250) — matrix operand order repaired and checked by reading, not compiled, not run on hardware
//
#include <hip/hip_runtime.h>
#include <hip/hip_bf16.h>
#include <math.h>

// Problem constants (from reference)
#define NN 256
#define TT 2048
#define FF 128
#define UNITS 128
#define K2 256          // 2*F, GEMM K dimension

typedef __attribute__((ext_vector_type(16))) _Float16 v16h;
typedef __attribute__((ext_vector_type(8)))  float    v8f;

// ---------------------------------------------------------------------------
// Kernel 1: hidden_size[f] = sum_{n,t} x[n,t,f]^2   (grid-stride, float4)
// Flat float4 index i4: f-base = 4*(i4 % 32). With blockDim=256 and
// grid-stride multiples of 32, each thread's f-base is fixed = 4*(tid%32).
// ---------------------------------------------------------------------------
__global__ void k_sumsq(const float4* __restrict__ x4, float* __restrict__ hidden) {
    const long total4 = (long)NN * TT * FF / 4;   // 16,777,216
    float a0 = 0.f, a1 = 0.f, a2 = 0.f, a3 = 0.f;
    const long stride = (long)gridDim.x * blockDim.x;
    for (long i = (long)blockIdx.x * blockDim.x + threadIdx.x; i < total4; i += stride) {
        float4 v = x4[i];
        a0 += v.x * v.x;  a1 += v.y * v.y;  a2 += v.z * v.z;  a3 += v.w * v.w;
    }
    __shared__ float sh[FF];
    if (threadIdx.x < FF) sh[threadIdx.x] = 0.f;
    __syncthreads();
    const int fb = 4 * (threadIdx.x & 31);
    atomicAdd(&sh[fb + 0], a0);
    atomicAdd(&sh[fb + 1], a1);
    atomicAdd(&sh[fb + 2], a2);
    atomicAdd(&sh[fb + 3], a3);
    __syncthreads();
    if (threadIdx.x < FF) atomicAdd(&hidden[threadIdx.x], sh[threadIdx.x]);
}

// ---------------------------------------------------------------------------
// Kernel 2: convert W (128 x 256 f32, row-major) to f16 workspace copy.
// ---------------------------------------------------------------------------
__global__ void k_convw(const float* __restrict__ W, _Float16* __restrict__ W16) {
    int i = blockIdx.x * blockDim.x + threadIdx.x;
    if (i < UNITS * K2) W16[i] = (_Float16)W[i];
}

// ---------------------------------------------------------------------------
// Async load of one 512B row tile (per-lane 16B) into this wave's LDS stage.
// VGLOBAL GV mode: global_load_async_to_lds_b128 vdst(LDS byte addr), v[addr64], off
// Tracked by ASYNCcnt. LDS byte offset = low 32 bits of the flat address of
// the __shared__ stage.
// ---------------------------------------------------------------------------
__device__ __forceinline__ void async_row_to_lds(const float* gsrc, void* lds_dst) {
    unsigned ldsoff = (unsigned)(unsigned long long)lds_dst;
    asm volatile("global_load_async_to_lds_b128 %0, %1, off"
                 :: "v"(ldsoff), "v"(gsrc)
                 : "memory");
}

// ---------------------------------------------------------------------------
// Kernel 3: per-row fused score -> online softmax -> context.
// One block (8 waves) per n. Each wave streams T/8 = 256 rows of x[n] through
// a private double-buffered LDS stage filled with global_load_async_to_lds_b128
// (ASYNCcnt pipeline, one tile always in flight; s_wait_asynccnt 0x1 gates the
// consumer). Score is a 128-length dot with v = h_t * hidden, reduced across
// the wave with __shfl_xor (wave32). Online softmax keeps m/l/context in regs.
// Writes the f16 pre-activation row [context | h_t] (256 halves) to P16.
// ---------------------------------------------------------------------------
#define WAVES 8
__global__ void k_attn(const float* __restrict__ x,
                       const float* __restrict__ hidden,
                       _Float16* __restrict__ P16) {
    const int n    = blockIdx.x;
    const int lane = threadIdx.x & 31;
    const int wave = threadIdx.x >> 5;
    const float* __restrict__ xn = x + (long)n * TT * FF;

    __shared__ float sh_v[FF];
    __shared__ float sh_h[FF];
    __shared__ float sh_m[WAVES];
    __shared__ float sh_l[WAVES];
    __shared__ float sh_c[WAVES][FF];
    __shared__ float stage[WAVES][2][FF];   // per-wave double buffer, 2 x 512 B

    if (threadIdx.x < FF) {
        float h = xn[(long)(TT - 1) * FF + threadIdx.x];
        sh_h[threadIdx.x] = h;
        sh_v[threadIdx.x] = h * hidden[threadIdx.x];
    }
    __syncthreads();

    const float4 v4 = *(const float4*)&sh_v[4 * lane];

    float m = -INFINITY, l = 0.f;
    float c0 = 0.f, c1 = 0.f, c2 = 0.f, c3 = 0.f;

    // Prologue: stage the first tile for this wave.
    async_row_to_lds(xn + (long)wave * FF + 4 * lane, &stage[wave][0][4 * lane]);

    int buf = 0;
    for (int t = wave; t < TT; t += WAVES) {
        if (t + WAVES < TT) {
            // Issue next tile, then wait until only that one is outstanding:
            // current tile is guaranteed complete (async loads retire in order).
            async_row_to_lds(xn + (long)(t + WAVES) * FF + 4 * lane,
                             &stage[wave][buf ^ 1][4 * lane]);
            asm volatile("s_wait_asynccnt 0x1" ::: "memory");
        } else {
            asm volatile("s_wait_asynccnt 0x0" ::: "memory");
        }

        float4 x4 = *(const float4*)&stage[wave][buf][4 * lane];

        float p = x4.x * v4.x + x4.y * v4.y + x4.z * v4.z + x4.w * v4.w;
        #pragma unroll
        for (int off = 16; off > 0; off >>= 1) p += __shfl_xor(p, off, 32);
        const float s = p;  // all lanes hold the score

        if (s > m) {
            float scale = __expf(m - s);   // first iter: exp(-inf)=0 zeroes accums
            l *= scale; c0 *= scale; c1 *= scale; c2 *= scale; c3 *= scale;
            m = s;
        }
        float e = __expf(s - m);
        l += e;
        c0 += e * x4.x;  c1 += e * x4.y;  c2 += e * x4.z;  c3 += e * x4.w;
        buf ^= 1;
    }

    if (lane == 0) { sh_m[wave] = m; sh_l[wave] = l; }
    sh_c[wave][4 * lane + 0] = c0;
    sh_c[wave][4 * lane + 1] = c1;
    sh_c[wave][4 * lane + 2] = c2;
    sh_c[wave][4 * lane + 3] = c3;
    __syncthreads();

    if (threadIdx.x < FF) {
        const int f = threadIdx.x;
        float M = sh_m[0];
        #pragma unroll
        for (int w = 1; w < WAVES; w++) M = fmaxf(M, sh_m[w]);
        float L = 0.f, C = 0.f;
        #pragma unroll
        for (int w = 0; w < WAVES; w++) {
            float sc = __expf(sh_m[w] - M);
            L += sh_l[w] * sc;
            C += sh_c[w][f] * sc;
        }
        float ctx = C / L;
        P16[(long)n * K2 + f]      = (_Float16)ctx;      // context half
        P16[(long)n * K2 + FF + f] = (_Float16)sh_h[f];  // h_t half
    }
}

// ---------------------------------------------------------------------------
// Kernel 4: out = tanh(P(256x256) @ W^T(256x128) + b) via v_wmma_f32_16x16x32_f16.
// One wave per 16x16 output tile: 16 M-tiles (blocks) x 8 N-tiles (waves).
// A (16x32 f16): lanes 0-15 M rows, halves h -> K = h
// + (laneHi?16:0). B (32x16 f16): lane&15 = N col, halves 0..15 -> K 0..15
// (lanes 0-15) / K 16..31 (lanes 16-31). Operands are L2-hot (<200 KB).
// ---------------------------------------------------------------------------
__global__ void k_gemm(const _Float16* __restrict__ P16,
                       const _Float16* __restrict__ W16,
                       const float* __restrict__ bias,
                       float* __restrict__ out) {
    const int lane = threadIdx.x & 31;
    const int mt   = blockIdx.x;        // 0..15
    const int nt   = threadIdx.x >> 5;  // 0..7

    const int  laneHi = (lane >= 16) ? 1 : 0;
    const int  mrow   = mt * 16 + (lane & 15);
    const int  ncol   = nt * 16 + (lane & 15);

    v8f acc = {};
    const _Float16* __restrict__ Arow = P16 + (long)mrow * K2;   // P row-major
    const _Float16* __restrict__ Bcol = W16 + (long)ncol * K2;   // B[k][ncol] = W16[ncol*K2+k]

    for (int kk = 0; kk < K2; kk += 32) {
        v16h a, b;
        #pragma unroll
        for (int h = 0; h < 16; h++) {
            int kA = h + (laneHi ? 16 : 0);
            a[h] = Arow[kk + kA];
        }
        const int kbB = laneHi ? 16 : 0;
        #pragma unroll
        for (int h = 0; h < 16; h++) {
            b[h] = Bcol[kk + kbB + h];
        }
        acc = __builtin_amdgcn_wmma_f32_16x16x32_f16(
            /*neg_a=*/false, a, /*neg_b=*/false, b,
            /*c_mod=*/(short)0, acc, /*reuse_a=*/false, /*reuse_b=*/false);
    }

    // D layout: VGPR r -> M = mt*16 + r + (laneHi?8:0), N = lane&15
    const float bv    = bias[ncol];
    const int   mbase = mt * 16 + (laneHi ? 8 : 0);
    #pragma unroll
    for (int r = 0; r < 8; r++) {
        float val = tanhf(acc[r] + bv);
        out[(long)(mbase + r) * UNITS + ncol] = val;
    }
}

// ---------------------------------------------------------------------------
extern "C" void kernel_launch(void* const* d_in, const int* in_sizes, int n_in,
                              void* d_out, int out_size, void* d_ws, size_t ws_size,
                              hipStream_t stream) {
    const float* x = (const float*)d_in[0];   // (256, 2048, 128) f32
    const float* W = (const float*)d_in[1];   // (128, 256) f32
    const float* b = (const float*)d_in[2];   // (128,) f32
    float* out = (float*)d_out;               // (256, 128) f32

    // Workspace layout
    float*     hidden = (float*)d_ws;                                   // 512 B
    _Float16*  P16    = (_Float16*)((char*)d_ws + 1024);                // 256*256*2 = 128 KB
    _Float16*  W16    = (_Float16*)((char*)d_ws + 1024 + NN * K2 * 2);  // 128*256*2 = 64 KB

    hipMemsetAsync(hidden, 0, FF * sizeof(float), stream);

    k_sumsq<<<1024, 256, 0, stream>>>((const float4*)x, hidden);
    k_convw<<<(UNITS * K2 + 255) / 256, 256, 0, stream>>>(W, W16);
    k_attn <<<NN, 32 * WAVES, 0, stream>>>(x, hidden, P16);
    k_gemm <<<NN / 16, 32 * 8, 0, stream>>>(P16, W16, b, out);
}
